// DHP_1314259992584
// MI455X (gfx1250) — matrix operand order repaired and checked by reading, not compiled, not run on hardware
//
#include <hip/hip_runtime.h>
#include <hip/hip_bf16.h>
#include <stdint.h>

// ---------------------------------------------------------------------------
// Types for CDNA5 WMMA (wave32): v_wmma_f32_16x16x32_bf16, and TDM descriptors
// ---------------------------------------------------------------------------
typedef __attribute__((ext_vector_type(16))) __bf16 v16bf;
typedef __attribute__((ext_vector_type(8)))  float  v8f;
typedef unsigned int v4u  __attribute__((ext_vector_type(4)));
typedef int          v8i  __attribute__((ext_vector_type(8)));
typedef int          v4i  __attribute__((ext_vector_type(4)));

#define DFEAT 128
#define KHYP  8

#if __has_builtin(__builtin_amdgcn_tensor_load_to_lds) && \
    __has_builtin(__builtin_amdgcn_s_wait_tensorcnt)
#define USE_TDM 1
#else
#define USE_TDM 0
#endif

__device__ __forceinline__ unsigned short f2bf(float x) {
    union { float f; uint32_t u; } c; c.f = x;
    uint32_t u = c.u;
    uint32_t r = (u + 0x7FFFu + ((u >> 16) & 1u)) >> 16;   // round-to-nearest-even
    return (unsigned short)r;
}

// B fragment for V_WMMA_F32_16X16X32_BF16.
// Lane supplies column n; K values = kb + 8*half + 0..7, kb + 16 + 8*half + 0..7, 2 packed per VGPR
// (low 16 bits = even K). Loaded strided from row-major W[k][n], ld = ldw.
__device__ __forceinline__ v16bf load_bfrag(const float* __restrict__ W,
                                            int ldw, int kb, int n, int half) {
    union { v16bf v; unsigned short s[16]; } u;
#pragma unroll
    for (int j = 0; j < 16; ++j)
        u.s[j] = f2bf(W[(size_t)(kb + 8 * half + j + (j >= 8 ? 8 : 0)) * ldw + n]);
    return u.v;
}

// A fragment from an LDS bf16 tile row (row = l16 of the 16-row tile).
// Lane holds M = l16; K runs: [kb+8*half, +8) and [kb+16+8*half, +8).
__device__ __forceinline__ v16bf load_afrag(const unsigned short* __restrict__ arow,
                                            int kb, int half) {
    union { v16bf v; uint4 u[2]; } ua;
    ua.u[0] = *(const uint4*)(arow + kb + 8 * half);
    ua.u[1] = *(const uint4*)(arow + kb + 16 + 8 * half);
    return ua.v;
}

#if USE_TDM
// Issue a TDM DMA of one [16 x 128] f32 row tile (rows rowBase..rowBase+15,
// OOB rows zero-filled via tensor_dim1 clamp) into LDS at byte offset ldsOff.
__device__ __forceinline__ void tdm_load_tile(const float* __restrict__ X,
                                              int Nrows, int rowBase,
                                              unsigned ldsOff) {
    const int rowsLeft = Nrows - rowBase;            // >= 1
    const unsigned long long ga =
        (unsigned long long)(const void*)(X + (size_t)rowBase * DFEAT);
    // D# group 0: count=1 | lds_addr | global_addr[56:0] | type=2
    v4u g0;
    g0.x = 0x1u;
    g0.y = ldsOff;
    g0.z = (unsigned)(ga & 0xFFFFFFFFull);
    g0.w = (unsigned)((ga >> 32) & 0x01FFFFFFull) | 0x80000000u;
    // D# group 1: data_size=4B; tensor = rowsLeft x 128; tile = 16 x 128;
    // dim0 stride = 128 elements.
    v8i g1;
    g1[0] = 0x00020000;                                   // data_size=2 (4B)
    g1[1] = (int)(128u << 16);                            // tensor_dim0 lo16
    g1[2] = (int)(((unsigned)rowsLeft & 0xFFFFu) << 16);  // dim0 hi | dim1 lo
    g1[3] = (int)((((unsigned)rowsLeft >> 16) & 0xFFFFu)
                  | (128u << 16));                        // dim1 hi | tile_dim0
    g1[4] = 16;                                           // tile_dim1 | tile_dim2=0
    g1[5] = 128;                                          // dim0_stride lo32
    g1[6] = 0;
    g1[7] = 0;
    v4i g2 = {0, 0, 0, 0};                   // 2-D tensor: groups 2/3 unused
    v4i g3 = {0, 0, 0, 0};
    v8i g4 = {0, 0, 0, 0, 0, 0, 0, 0};       // extended/VADDR4 slot: unused
    __builtin_amdgcn_tensor_load_to_lds(g0, g1, g2, g3, g4, 0);
}
#endif

// ---------------------------------------------------------------------------
// Kernel 1: Out[N,128] = act(X)[N,128] @ W[128,128]  (act = optional ReLU)
// Block: 256 thr = 8 waves; wave w owns output columns [16w, 16w+16).
// Double-buffered TDM pipeline: the next tile's DMA runs under the current
// tile's convert + WMMA chain (TDM ops from one wave complete in order, so
// s_wait_tensorcnt 0 at loop top == "current tile landed").
// ---------------------------------------------------------------------------
__global__ __launch_bounds__(256) void gemm_xw_kernel(
    const float* __restrict__ X, const float* __restrict__ W,
    float* __restrict__ Out, int Nrows, int relu_in)
{
#if USE_TDM
    __shared__ float          Afs[2][16 * DFEAT];    // 2 x 8 KB TDM staging (f32)
#endif
    __shared__ unsigned short Atile[16][DFEAT];      // 4 KB bf16 WMMA tile

    const int tid  = threadIdx.x;
    const int wave = tid >> 5;
    const int lane = tid & 31;
    const int half = lane >> 4;
    const int l16  = lane & 15;
    const int col  = wave * 16 + l16;

    v16bf bfrag[4];
#pragma unroll
    for (int c = 0; c < 4; ++c)
        bfrag[c] = load_bfrag(W, DFEAT, c * 32, col, half);

    const int numTiles = (Nrows + 15) >> 4;

#if USE_TDM
    // LDS byte addresses: flat->LDS mapping keeps the offset in addr[31:0].
    unsigned ldsOff[2];
    ldsOff[0] = (unsigned)(size_t)(const void*)&Afs[0][0];
    ldsOff[1] = (unsigned)(size_t)(const void*)&Afs[1][0];
    if (wave == 0 && blockIdx.x < (unsigned)numTiles)
        tdm_load_tile(X, Nrows, blockIdx.x << 4, ldsOff[0]);   // prologue
    int buf = 0;
#endif

    for (int t = blockIdx.x; t < numTiles; t += gridDim.x) {
        const int rowBase  = t << 4;
        const int rowsLeft = Nrows - rowBase;        // >= 1

#if USE_TDM
        if (wave == 0) {
            __builtin_amdgcn_s_wait_tensorcnt(0);    // current tile landed
            int tn = t + gridDim.x;
            if (tn < numTiles)                        // prefetch next tile's DMA
                tdm_load_tile(X, Nrows, tn << 4, ldsOff[buf ^ 1]);
        }
        __syncthreads();                             // staging tile visible
        const float* src = &Afs[buf][0];
        for (int i = tid; i < 16 * DFEAT; i += 256) {
            float v = src[i];
            if (relu_in) v = fmaxf(v, 0.f);
            Atile[i >> 7][i & (DFEAT - 1)] = f2bf(v);
        }
        buf ^= 1;
#else
        __syncthreads();
        for (int i = tid; i < 16 * DFEAT; i += 256) {
            int r = i >> 7, k = i & (DFEAT - 1);
            int row = rowBase + r;
            float v = (row < Nrows) ? X[(size_t)row * DFEAT + k] : 0.f;
            if (relu_in) v = fmaxf(v, 0.f);
            Atile[r][k] = f2bf(v);
        }
#endif
        __syncthreads();

        // Load all four A fragments, then run the WMMA chain.
        const unsigned short* arow = &Atile[l16][0];
        v16bf afr[4];
#pragma unroll
        for (int c = 0; c < 4; ++c) afr[c] = load_afrag(arow, c * 32, half);
        v8f acc = {};
#pragma unroll
        for (int c = 0; c < 4; ++c)
            acc = __builtin_amdgcn_wmma_f32_16x16x32_bf16(
                false, afr[c], false, bfrag[c], (short)0, acc, false, false);

        if (rowsLeft >= 16) {                        // fast path: full tile
            float* outp = Out + (size_t)(rowBase + 8 * half) * DFEAT + col;
#pragma unroll
            for (int vv = 0; vv < 8; ++vv) outp[(size_t)vv * DFEAT] = acc[vv];
        } else {
#pragma unroll
            for (int vv = 0; vv < 8; ++vv) {
                int row = rowBase + vv + 8 * half;
                if (row < Nrows) Out[(size_t)row * DFEAT + col] = acc[vv];
            }
        }
        __syncthreads();                             // Atile/staging reusable
    }
}

// ---------------------------------------------------------------------------
// Kernel 2: Y[i,f] = bias[f]   (segment-sum accumulator init)
// ---------------------------------------------------------------------------
__global__ void bias_init_kernel(float* __restrict__ Y,
                                 const float* __restrict__ b, int total) {
    int i = blockIdx.x * blockDim.x + threadIdx.x;
    if (i < total) Y[i] = b[i & (DFEAT - 1)];
}

// ---------------------------------------------------------------------------
// Kernel 3: SpMM scatter:  Y[row[e], :] += val[e] * Sup[col[e], :]
// One wave per edge, one float4 (4 features) per lane.
// ---------------------------------------------------------------------------
__global__ __launch_bounds__(256) void spmm_atomic_kernel(
    const float* __restrict__ Sup, const int* __restrict__ erow,
    const int* __restrict__ ecol, const float* __restrict__ eval,
    float* __restrict__ Y, int E)
{
    int e = (blockIdx.x * blockDim.x + threadIdx.x) >> 5;
    if (e >= E) return;
    int lane = threadIdx.x & 31;
    int r = erow[e], c = ecol[e];
    float v = eval[e];
    float4 m = ((const float4*)(Sup + (size_t)c * DFEAT))[lane];
    float* y = Y + (size_t)r * DFEAT + lane * 4;
    atomicAdd(y + 0, v * m.x);
    atomicAdd(y + 1, v * m.y);
    atomicAdd(y + 2, v * m.z);
    atomicAdd(y + 3, v * m.w);
}

// ---------------------------------------------------------------------------
// Kernel 4: inner_prob — gather-mean pooling + [16x128]@[128x256] WMMA MLP
// + ReLU + p_w dot + stable BCE, accumulated into *loss_out.
// Block: 256 thr = 8 waves; wave w owns hidden columns [32w, 32w+32) (2 tiles).
// ---------------------------------------------------------------------------
__global__ __launch_bounds__(256) void inner_prob_kernel(
    const float* __restrict__ Xn, const int* __restrict__ pos,
    const int* __restrict__ neg, const float* __restrict__ h_w,
    const float* __restrict__ h_b, const float* __restrict__ p_w,
    const float* __restrict__ p_b, float* __restrict__ loss_out,
    int S, int TOT, int NH)
{
    __shared__ unsigned short Atile[16][DFEAT];      // pooled features, bf16
    __shared__ float logits[16];

    const int tid  = threadIdx.x;
    const int wave = tid >> 5;
    const int lane = tid & 31;
    const int half = lane >> 4;
    const int l16  = lane & 15;
    const float pb = p_b[0];
    const float invTot = 1.0f / (float)TOT;

    // Hoist B fragments of h_w for this wave's two column tiles (registers).
    v16bf bfrag[2][4];
    float hb[2], pw[2];
#pragma unroll
    for (int ctp = 0; ctp < 2; ++ctp) {
        int n = (wave * 2 + ctp) * 16 + l16;
#pragma unroll
        for (int c = 0; c < 4; ++c)
            bfrag[ctp][c] = load_bfrag(h_w, NH, c * 32, n, half);
        hb[ctp] = h_b[n];
        pw[ctp] = p_w[n];
    }

    const int numTiles = (TOT + 15) >> 4;
    for (int t = blockIdx.x; t < numTiles; t += gridDim.x) {
        const int sBase = t << 4;
        __syncthreads();
        // Pooling: mean of 8 gathered node rows per sample, f32 -> bf16 tile.
        for (int i = tid; i < 16 * DFEAT; i += 256) {
            int si = i >> 7, f = i & (DFEAT - 1);
            int s = sBase + si;
            float acc = 0.f;
            if (s < TOT) {
                const int* nodes = (s < S) ? (pos + (size_t)s * KHYP)
                                           : (neg + (size_t)(s - S) * KHYP);
#pragma unroll
                for (int j = 0; j < KHYP; ++j)
                    acc += Xn[(size_t)nodes[j] * DFEAT + f];
                acc *= (1.0f / (float)KHYP);
            }
            Atile[si][f] = f2bf(acc);
        }
        if (tid < 16) logits[tid] = 0.f;
        __syncthreads();

        // Load the four shared A fragments once; reuse for both column tiles.
        const unsigned short* arow = &Atile[l16][0];
        v16bf afr[4];
#pragma unroll
        for (int c = 0; c < 4; ++c) afr[c] = load_afrag(arow, c * 32, half);

#pragma unroll
        for (int ctp = 0; ctp < 2; ++ctp) {
            v8f acc = {};
#pragma unroll
            for (int c = 0; c < 4; ++c)
                acc = __builtin_amdgcn_wmma_f32_16x16x32_bf16(
                    false, afr[c], false, bfrag[ctp][c], (short)0, acc, false, false);
            // hidden = relu(acc + h_b[n]); logit[M] += hidden * p_w[n]
#pragma unroll
            for (int vv = 0; vv < 8; ++vv) {
                float h = fmaxf(acc[vv] + hb[ctp], 0.f);
                atomicAdd(&logits[vv + 8 * half], h * pw[ctp]);
            }
        }
        __syncthreads();

        if (tid < 16) {
            int s = sBase + tid;
            if (s < TOT) {
                float z = logits[tid] + pb;
                float t1 = log1pf(expf(-fabsf(z)));
                float ls_pos = fminf(z, 0.f) - t1;    // log sigmoid(z)
                float ls_neg = fminf(-z, 0.f) - t1;   // log sigmoid(-z)
                float bce = (s < S) ? -ls_pos : -ls_neg;
                atomicAdd(loss_out, bce * invTot);    // LOSS_DECAY == 1
            }
        }
    }
}

// ---------------------------------------------------------------------------
// Kernel 5: out = emb + 0.5*relu(x1) + x2/3
// ---------------------------------------------------------------------------
__global__ void combine_kernel(const float* __restrict__ emb,
                               const float* __restrict__ x1,
                               const float* __restrict__ x2,
                               float* __restrict__ out, int total) {
    int i = blockIdx.x * blockDim.x + threadIdx.x;
    if (i < total)
        out[i] = emb[i] + 0.5f * fmaxf(x1[i], 0.f) + x2[i] * (1.0f / 3.0f);
}

__global__ void zero_scalar_kernel(float* __restrict__ p) { *p = 0.f; }

// ---------------------------------------------------------------------------
extern "C" void kernel_launch(void* const* d_in, const int* in_sizes, int n_in,
                              void* d_out, int out_size, void* d_ws, size_t ws_size,
                              hipStream_t stream) {
    const int*   edge_row = (const int*)  d_in[0];
    const int*   edge_col = (const int*)  d_in[1];
    const float* edge_val = (const float*)d_in[2];
    const int*   pos      = (const int*)  d_in[3];
    const int*   neg      = (const int*)  d_in[4];
    const float* emb      = (const float*)d_in[5];
    const float* gc1_w    = (const float*)d_in[6];
    const float* gc1_b    = (const float*)d_in[7];
    const float* gc2_w    = (const float*)d_in[8];
    const float* gc2_b    = (const float*)d_in[9];
    const float* h_w      = (const float*)d_in[10];
    const float* h_b      = (const float*)d_in[11];
    const float* p_w      = (const float*)d_in[12];
    const float* p_b      = (const float*)d_in[13];

    const int E   = in_sizes[0];
    const int S   = in_sizes[3] / KHYP;
    const int TOT = S + in_sizes[4] / KHYP;
    const int N   = in_sizes[5] / DFEAT;
    const int NH  = in_sizes[11];
    const size_t ND = (size_t)N * DFEAT;

    float* out_mat = (float*)d_out;
    float* loss    = out_mat + ND;

    float* support = (float*)d_ws;     // [N,128]
    float* x1      = support + ND;     // [N,128]
    float* x2      = x1 + ND;          // [N,128]

    const int gemmTiles = (N + 15) >> 4;
    const int gemmBlks  = gemmTiles < 1024 ? gemmTiles : 1024;  // >1 tile/block
    const int ipTiles   = (TOT + 15) >> 4;
    const int elemBlks  = (int)((ND + 255) / 256);
    const int spmmBlks  = (E + 7) / 8;           // 8 edges (waves) per block

    zero_scalar_kernel<<<1, 1, 0, stream>>>(loss);

    // loss += inner_prob(x0 = emb)
    inner_prob_kernel<<<ipTiles, 256, 0, stream>>>(emb, pos, neg, h_w, h_b,
                                                   p_w, p_b, loss, S, TOT, NH);
    // x1 = gcn(emb, gc1)
    gemm_xw_kernel<<<gemmBlks, 256, 0, stream>>>(emb, gc1_w, support, N, 0);
    bias_init_kernel<<<elemBlks, 256, 0, stream>>>(x1, gc1_b, (int)ND);
    spmm_atomic_kernel<<<spmmBlks, 256, 0, stream>>>(support, edge_row, edge_col,
                                                     edge_val, x1, E);
    // loss += inner_prob(x1)   (pre-ReLU, as in reference)
    inner_prob_kernel<<<ipTiles, 256, 0, stream>>>(x1, pos, neg, h_w, h_b,
                                                   p_w, p_b, loss, S, TOT, NH);
    // x2 = gcn(relu(x1), gc2)
    gemm_xw_kernel<<<gemmBlks, 256, 0, stream>>>(x1, gc2_w, support, N, 1);
    bias_init_kernel<<<elemBlks, 256, 0, stream>>>(x2, gc2_b, (int)ND);
    spmm_atomic_kernel<<<spmmBlks, 256, 0, stream>>>(support, edge_row, edge_col,
                                                     edge_val, x2, E);
    // loss += inner_prob(x2)
    inner_prob_kernel<<<ipTiles, 256, 0, stream>>>(x2, pos, neg, h_w, h_b,
                                                   p_w, p_b, loss, S, TOT, NH);
    // out = emb + 0.5*relu(x1) + x2/3
    combine_kernel<<<elemBlks, 256, 0, stream>>>(emb, x1, x2, out_mat, (int)ND);
}
